// EarlyExitResNetBKBlock_48421461295387
// MI455X (gfx1250) — matrix operand order repaired and checked by reading, not compiled, not run on hardware
//
#include <hip/hip_runtime.h>
#include <hip/hip_bf16.h>

// Problem constants (reference: B=2,N=2048,D=1024,E=4,K=1,F=4096,V=32000)
#define TT  4096   // tokens = B*N
#define DD  1024
#define EE  4
#define FF  4096
#define VV  32000
#define KCH 128            // K-chunk staged in LDS (256B of bf16 per row)
#define PAD 8              // 16B pad per row -> LDS row pitch 272B (bank spread)
#define LDA (KCH + PAD)

static_assert(DD % KCH == 0 && FF % KCH == 0, "K chunking");
static_assert(FF % 64 == 0 && DD % 64 == 0 && VV % 64 == 0 && TT % 16 == 0, "tiling");

typedef __attribute__((ext_vector_type(16))) __bf16 bf16x16;
typedef __attribute__((ext_vector_type(8)))  float  f32x8;
typedef __attribute__((ext_vector_type(4)))  unsigned int u32x4;
typedef __attribute__((ext_vector_type(8)))  int i32x8;
typedef __attribute__((ext_vector_type(4)))  int i32x4;

// ---- WMMA fragment loaders (contiguous k = 16*(lane/16) + i, wave32) ----

// A-matrix 16x32 bf16 tile from LDS (row-major, pitch LDA).
// lanes 0-15: row=lane, K={0..15}; lanes 16-31: row=lane-16, K={16..31}
__device__ __forceinline__ bf16x16 load_A_lds(const __bf16* As, int lane) {
  int row  = lane & 15;
  int koff = (lane >> 4) << 4;
  bf16x16 a;
#pragma unroll
  for (int i = 0; i < 16; ++i) {
    int k = koff + i;
    a[i] = As[row * LDA + k];
  }
  return a;
}

// B-matrix 32x16 tile from row-major f32 global (leading dim ldb), cvt->bf16.
__device__ __forceinline__ bf16x16 load_B_f32(const float* B, int ldb, int lane) {
  int col  = lane & 15;
  int koff = (lane >> 4) << 4;
  const float* p = B + (size_t)koff * (size_t)ldb + col;
  bf16x16 b;
#pragma unroll
  for (int j = 0; j < 8; ++j) {
    b[2 * j]     = (__bf16)p[0];
    b[2 * j + 1] = (__bf16)p[ldb];
    p += 2 * (size_t)ldb;
  }
  return b;
}

// ---- TDM gather-mode descriptor: DMA 16 gathered bf16 rows (KCH elems each)
// from global into LDS, HW-padded to the LDA pitch. Issued per-wave; TENSORcnt. ----
__device__ __forceinline__ i32x4 pack_idx(const int* toks, int base) {
  i32x4 v;
#pragma unroll
  for (int j = 0; j < 4; ++j) {
    int a = toks[base + 2 * j];     if (a < 0) a = 0;   // clamp: garbage rows masked at store
    int b = toks[base + 2 * j + 1]; if (b < 0) b = 0;
    v[j] = __builtin_amdgcn_readfirstlane((a & 0xFFFF) | (b << 16));
  }
  return v;
}

__device__ __forceinline__ void tdm_gather16_bf16(unsigned lds_addr, const __bf16* g,
                                                  unsigned stride_elems, i32x4 gi0, i32x4 gi1) {
  unsigned long long ga = (unsigned long long)(uintptr_t)g;
  u32x4 g0;
  g0[0] = 1u | (1u << 31);                                   // count=1, gather_mode, 16-bit idx
  g0[1] = (unsigned)__builtin_amdgcn_readfirstlane((int)lds_addr);
  g0[2] = (unsigned)__builtin_amdgcn_readfirstlane((int)(unsigned)ga);
  g0[3] = (unsigned)__builtin_amdgcn_readfirstlane(
              (int)(((unsigned)(ga >> 32) & 0x01FFFFFFu) | (2u << 30)));  // addr[56:32], type=2
  i32x8 g1;
  // data_size=1(2B), pad_enable, pad_interval=5(64dw=256B), pad_amount=3(4dw=16B)
  g1[0] = (int)((1u << 16) | (1u << 20) | (5u << 22) | (3u << 25));
  g1[1] = (int)((KCH & 0xFFFFu) << 16);                      // tensor_dim0 = KCH   [63:48]
  g1[2] = (int)((TT & 0xFFFFu) << 16);                       // tensor_dim1 = TT    [95:80]
  g1[3] = (int)((KCH & 0xFFFFu) << 16);                      // tile_dim0 = KCH     [127:112]
  g1[4] = 16;                                                // tile_dim1 = 16 gather indices
  g1[5] = __builtin_amdgcn_readfirstlane((int)stride_elems); // tensor_dim0_stride  [191:160]
  g1[6] = 0;
  g1[7] = 0;
#if __clang_major__ >= 23
  i32x8 zz = {};
  __builtin_amdgcn_tensor_load_to_lds(g0, g1, gi0, gi1, zz, 0);
#else
  __builtin_amdgcn_tensor_load_to_lds(g0, g1, gi0, gi1, 0);
#endif
}

__device__ __forceinline__ unsigned lds_off(const void* p) {
  return (unsigned)(uintptr_t)p;   // flat LDS address: low 32 bits are the LDS byte offset
}

// ---- Kernel 1: LN1 + gate logits + top-1 routing (K=1 => weight exactly 1.0) ----
__global__ void ln1_gate_kernel(const float* __restrict__ x, const float* __restrict__ g,
                                const float* __restrict__ bb, const float* __restrict__ Wg,
                                __bf16* __restrict__ xn, int* __restrict__ counts,
                                int* __restrict__ lists) {
  int t = blockIdx.x;
  int tid = threadIdx.x;
  const float* xr = x + (size_t)t * DD;
  __shared__ float red[256];
  __shared__ float gr[EE][256];

  float s = 0.f;
  for (int d = tid; d < DD; d += 256) s += xr[d];
  red[tid] = s; __syncthreads();
  for (int off = 128; off > 0; off >>= 1) { if (tid < off) red[tid] += red[tid + off]; __syncthreads(); }
  float mu = red[0] * (1.f / DD);
  __syncthreads();

  float v = 0.f;
  for (int d = tid; d < DD; d += 256) { float dv = xr[d] - mu; v += dv * dv; }
  red[tid] = v; __syncthreads();
  for (int off = 128; off > 0; off >>= 1) { if (tid < off) red[tid] += red[tid + off]; __syncthreads(); }
  float rstd = rsqrtf(red[0] * (1.f / DD) + 1e-5f);

  float ga[EE] = {0.f, 0.f, 0.f, 0.f};
  for (int d = tid; d < DD; d += 256) {
    float xv = (xr[d] - mu) * rstd * g[d] + bb[d];
    xn[(size_t)t * DD + d] = (__bf16)xv;
#pragma unroll
    for (int e = 0; e < EE; ++e) ga[e] += xv * Wg[d * EE + e];
  }
#pragma unroll
  for (int e = 0; e < EE; ++e) gr[e][tid] = ga[e];
  __syncthreads();
  for (int off = 128; off > 0; off >>= 1) {
    if (tid < off) {
#pragma unroll
      for (int e = 0; e < EE; ++e) gr[e][tid] += gr[e][tid + off];
    }
    __syncthreads();
  }
  if (tid == 0) {
    int best = 0; float bv = gr[0][0];
#pragma unroll
    for (int e = 1; e < EE; ++e) { if (gr[e][0] > bv) { bv = gr[e][0]; best = e; } }  // first-max on ties
    int pos = atomicAdd(&counts[best], 1);
    lists[best * TT + pos] = t;
  }
}

// ---- Kernel 2: routed expert FFN stage 1: H = gelu(xn @ W1[e] + b1[e]) ----
__global__ void ffn1_kernel(const __bf16* __restrict__ xn, const float* __restrict__ W1,
                            const float* __restrict__ b1, const int* __restrict__ counts,
                            const int* __restrict__ lists, __bf16* __restrict__ H) {
  int e = blockIdx.z;
  int cnt = counts[e];
  int tbase = blockIdx.y * 16;
  if (tbase >= cnt) return;                 // block-uniform exit, EXEC stays full
  int tid = threadIdx.x, lane = tid & 31, w = tid >> 5;
  int cb = blockIdx.x * 64 + w * 16;

  __shared__ __bf16 As[2][16][LDA];
  __shared__ int toks[16];
  if (tid < 16) {
    int r = tbase + tid;
    toks[tid] = (r < cnt) ? lists[e * TT + r] : -1;
  }
  __syncthreads();
  i32x4 gi0 = pack_idx(toks, 0);
  i32x4 gi1 = pack_idx(toks, 8);

  const float* Wb = W1 + (size_t)e * DD * FF;
  f32x8 acc = {};
  if (w == 0) tdm_gather16_bf16(lds_off(&As[0][0][0]), xn, DD, gi0, gi1);
  int buf = 0;
  for (int kc = 0; kc < DD; kc += KCH) {
    if (w == 0) __builtin_amdgcn_s_wait_tensorcnt(0);
    __syncthreads();                        // staged chunk visible to all waves
    if (w == 0 && kc + KCH < DD)            // async prefetch of next chunk into other buffer
      tdm_gather16_bf16(lds_off(&As[buf ^ 1][0][0]), xn + (size_t)(kc + KCH), DD, gi0, gi1);
#pragma unroll
    for (int ks = 0; ks < KCH; ks += 32) {
      bf16x16 a  = load_A_lds(&As[buf][0][ks], lane);
      bf16x16 bf = load_B_f32(Wb + (size_t)(kc + ks) * FF + cb, FF, lane);
      acc = __builtin_amdgcn_wmma_f32_16x16x32_bf16(false, a, false, bf, (short)0, acc, false, false);
    }
    __syncthreads();                        // all waves done with buf before it is refilled
    buf ^= 1;
  }
  int col = lane & 15, rb = (lane >> 4) * 8;
  int f = cb + col;
  float bias = b1[e * FF + f];
#pragma unroll
  for (int vv = 0; vv < 8; ++vv) {
    int tk = toks[rb + vv];
    if (tk < 0) continue;
    float h = acc[vv] + bias;
    float gel = 0.5f * h * (1.f + tanhf(0.7978845608028654f * (h + 0.044715f * h * h * h)));
    H[(size_t)tk * FF + f] = (__bf16)gel;
  }
}

// ---- Kernel 3: routed expert FFN stage 2 + residual: xproc = x + H @ W2[e] + b2[e] ----
__global__ void ffn2_kernel(const __bf16* __restrict__ H, const float* __restrict__ W2,
                            const float* __restrict__ b2, const int* __restrict__ counts,
                            const int* __restrict__ lists, const float* __restrict__ x,
                            float* __restrict__ xproc) {
  int e = blockIdx.z;
  int cnt = counts[e];
  int tbase = blockIdx.y * 16;
  if (tbase >= cnt) return;
  int tid = threadIdx.x, lane = tid & 31, w = tid >> 5;
  int cb = blockIdx.x * 64 + w * 16;

  __shared__ __bf16 As[2][16][LDA];
  __shared__ int toks[16];
  if (tid < 16) {
    int r = tbase + tid;
    toks[tid] = (r < cnt) ? lists[e * TT + r] : -1;
  }
  __syncthreads();
  i32x4 gi0 = pack_idx(toks, 0);
  i32x4 gi1 = pack_idx(toks, 8);

  const float* Wb = W2 + (size_t)e * FF * DD;
  f32x8 acc = {};
  if (w == 0) tdm_gather16_bf16(lds_off(&As[0][0][0]), H, FF, gi0, gi1);
  int buf = 0;
  for (int kc = 0; kc < FF; kc += KCH) {
    if (w == 0) __builtin_amdgcn_s_wait_tensorcnt(0);
    __syncthreads();
    if (w == 0 && kc + KCH < FF)
      tdm_gather16_bf16(lds_off(&As[buf ^ 1][0][0]), H + (size_t)(kc + KCH), FF, gi0, gi1);
#pragma unroll
    for (int ks = 0; ks < KCH; ks += 32) {
      bf16x16 a  = load_A_lds(&As[buf][0][ks], lane);
      bf16x16 bf = load_B_f32(Wb + (size_t)(kc + ks) * DD + cb, DD, lane);
      acc = __builtin_amdgcn_wmma_f32_16x16x32_bf16(false, a, false, bf, (short)0, acc, false, false);
    }
    __syncthreads();
    buf ^= 1;
  }
  int col = lane & 15, rb = (lane >> 4) * 8;
  int dcol = cb + col;
  float bias = b2[e * DD + dcol];
#pragma unroll
  for (int vv = 0; vv < 8; ++vv) {
    int tk = toks[rb + vv];
    if (tk < 0) continue;
    xproc[(size_t)tk * DD + dcol] = x[(size_t)tk * DD + dcol] + acc[vv] + bias;
  }
}

// ---- Kernel 4: LN2 -> xe (bf16) ----
__global__ void ln2_kernel(const float* __restrict__ xp, const float* __restrict__ g,
                           const float* __restrict__ bb, __bf16* __restrict__ xe) {
  int t = blockIdx.x;
  int tid = threadIdx.x;
  const float* xr = xp + (size_t)t * DD;
  __shared__ float red[256];
  float s = 0.f;
  for (int d = tid; d < DD; d += 256) s += xr[d];
  red[tid] = s; __syncthreads();
  for (int off = 128; off > 0; off >>= 1) { if (tid < off) red[tid] += red[tid + off]; __syncthreads(); }
  float mu = red[0] * (1.f / DD);
  __syncthreads();
  float v = 0.f;
  for (int d = tid; d < DD; d += 256) { float dv = xr[d] - mu; v += dv * dv; }
  red[tid] = v; __syncthreads();
  for (int off = 128; off > 0; off >>= 1) { if (tid < off) red[tid] += red[tid + off]; __syncthreads(); }
  float rstd = rsqrtf(red[0] * (1.f / DD) + 1e-5f);
  for (int d = tid; d < DD; d += 256)
    xe[(size_t)t * DD + d] = (__bf16)((xr[d] - mu) * rstd * g[d] + bb[d]);
}

// ---- Kernel 5: exit logits = xe @ Wout + bout (HBM-write-bound; bf16 WMMA) ----
__global__ void vocab_kernel(const __bf16* __restrict__ xe, const float* __restrict__ Wout,
                             const float* __restrict__ bout, float* __restrict__ logits) {
  int tbase = blockIdx.y * 16;
  int tid = threadIdx.x, lane = tid & 31, w = tid >> 5;
  int cb = blockIdx.x * 64 + w * 16;

  __shared__ __bf16 As[2][16][LDA];
  __shared__ int toks[16];
  if (tid < 16) toks[tid] = tbase + tid;    // contiguous rows via the same gather path
  __syncthreads();
  i32x4 gi0 = pack_idx(toks, 0);
  i32x4 gi1 = pack_idx(toks, 8);

  f32x8 acc = {};
  if (w == 0) tdm_gather16_bf16(lds_off(&As[0][0][0]), xe, DD, gi0, gi1);
  int buf = 0;
  for (int kc = 0; kc < DD; kc += KCH) {
    if (w == 0) __builtin_amdgcn_s_wait_tensorcnt(0);
    __syncthreads();
    if (w == 0 && kc + KCH < DD)
      tdm_gather16_bf16(lds_off(&As[buf ^ 1][0][0]), xe + (size_t)(kc + KCH), DD, gi0, gi1);
    if (kc + KCH < DD)  // hint next Wout chunk toward the caches (global_prefetch_b8)
      __builtin_prefetch(Wout + (size_t)(kc + KCH) * VV + cb + lane, 0, 1);
#pragma unroll
    for (int ks = 0; ks < KCH; ks += 32) {
      bf16x16 a  = load_A_lds(&As[buf][0][ks], lane);
      bf16x16 bf = load_B_f32(Wout + (size_t)(kc + ks) * VV + cb, VV, lane);
      acc = __builtin_amdgcn_wmma_f32_16x16x32_bf16(false, a, false, bf, (short)0, acc, false, false);
    }
    __syncthreads();
    buf ^= 1;
  }
  int col = lane & 15, rb = (lane >> 4) * 8;
  int vcol = cb + col;
  float bias = bout[vcol];
#pragma unroll
  for (int vv = 0; vv < 8; ++vv) {
    int t = tbase + rb + vv;
    logits[(size_t)t * VV + vcol] = acc[vv] + bias;
  }
}

extern "C" void kernel_launch(void* const* d_in, const int* in_sizes, int n_in,
                              void* d_out, int out_size, void* d_ws, size_t ws_size,
                              hipStream_t stream) {
  const float* x     = (const float*)d_in[0];
  const float* ln1_g = (const float*)d_in[1];
  const float* ln1_b = (const float*)d_in[2];
  const float* Wg    = (const float*)d_in[3];
  const float* W1    = (const float*)d_in[4];
  const float* b1    = (const float*)d_in[5];
  const float* W2    = (const float*)d_in[6];
  const float* b2    = (const float*)d_in[7];
  const float* ln2_g = (const float*)d_in[8];
  const float* ln2_b = (const float*)d_in[9];
  const float* Wout  = (const float*)d_in[10];
  const float* bout  = (const float*)d_in[11];

  float* xproc  = (float*)d_out;                       // (B,N,D) output 0
  float* logits = xproc + (size_t)TT * DD;             // (B,N,V) output 1

  // Workspace layout (~50.5 MB), fully rewritten every call.
  char* ws = (char*)d_ws;
  __bf16* xn = (__bf16*)ws;  ws += (size_t)TT * DD * sizeof(__bf16);
  __bf16* xe = (__bf16*)ws;  ws += (size_t)TT * DD * sizeof(__bf16);
  __bf16* H  = (__bf16*)ws;  ws += (size_t)TT * FF * sizeof(__bf16);
  int* counts = (int*)ws;    ws += 256;
  int* lists  = (int*)ws;    ws += (size_t)EE * TT * sizeof(int);

  hipMemsetAsync(counts, 0, EE * sizeof(int), stream);
  ln1_gate_kernel<<<TT, 256, 0, stream>>>(x, ln1_g, ln1_b, Wg, xn, counts, lists);
  ffn1_kernel<<<dim3(FF / 64, TT / 16, EE), 128, 0, stream>>>(xn, W1, b1, counts, lists, H);
  ffn2_kernel<<<dim3(DD / 64, TT / 16, EE), 128, 0, stream>>>(H, W2, b2, counts, lists, x, xproc);
  ln2_kernel<<<TT, 256, 0, stream>>>(xproc, ln2_g, ln2_b, xe);
  vocab_kernel<<<dim3(VV / 64, TT / 16), 128, 0, stream>>>(xe, Wout, bout, logits);
}
